// obj_Discriminator_6459630813307
// MI455X (gfx1250) — compile-verified, matrix operand order checked by reading, not run on hardware
//
#include <hip/hip_runtime.h>
#include <hip/hip_bf16.h>
#include <math.h>

typedef __attribute__((ext_vector_type(16))) _Float16 v16h;
typedef __attribute__((ext_vector_type(8)))  float    v8f;
typedef __attribute__((ext_vector_type(4)))  unsigned u32x4;
typedef __attribute__((ext_vector_type(8)))  int      i32x8;
typedef __attribute__((ext_vector_type(4)))  int      i32x4;

union FragH { v16h h; unsigned u[8]; };

// ---------------------------------------------------------------------------
// Tensor Data Mover (CDNA5): build D# per CDNA5 ISA and issue.
// group0: [1:0]=count(1), [63:32]=lds_addr, [120:64]=global_addr, [127:126]=2
// group1: [17:16]=data_size(1 => 2B), dim0[79:48], dim1[111:80],
//         tile0[127:112], tile1[143:128], dim0_stride[207:160]
// ---------------------------------------------------------------------------
#if defined(__has_builtin)
#if __has_builtin(__builtin_amdgcn_tensor_load_to_lds) && \
    __has_builtin(__builtin_amdgcn_tensor_store_from_lds) && \
    __has_builtin(__builtin_amdgcn_s_wait_tensorcnt)
#define USE_TDM 1
#endif
#endif
#ifndef USE_TDM
#define USE_TDM 0
#endif

__device__ __forceinline__ unsigned lds_off(const void* p) {
  // flat shared addr: [63:32]=aperture, [31:0]=LDS byte offset
  return (unsigned)(uintptr_t)p;
}

#if USE_TDM
__device__ __forceinline__ void tdm_build(u32x4& g0, i32x8& g1,
                                          unsigned ldsoff, unsigned long long ga,
                                          unsigned long long tw, unsigned th,
                                          unsigned long long stride,
                                          unsigned tilew, unsigned tileh) {
  g0[0] = 1u;                                       // count=1
  g0[1] = ldsoff;                                   // lds_addr
  g0[2] = (unsigned)ga;                             // global_addr lo
  g0[3] = (unsigned)((ga >> 32) & 0x01ffffffu) | (2u << 30);  // hi | type=2
  g1[0] = 0x10000;                                  // data_size = 1 (2 bytes)
  g1[1] = (int)((unsigned)(tw & 0xffffu) << 16);
  g1[2] = (int)(((unsigned)(tw >> 16) & 0xffffu) | ((th & 0xffffu) << 16));
  g1[3] = (int)(((th >> 16) & 0xffffu) | ((unsigned)tilew << 16));
  g1[4] = (int)(tileh & 0xffffu);
  g1[5] = (int)(stride & 0xffffffffull);
  g1[6] = (int)((stride >> 32) & 0xffffull);
  g1[7] = 0;
}

__device__ __forceinline__ void tdm_load_2d(unsigned ldsoff, const void* gptr,
                                            unsigned long long tw, unsigned th,
                                            unsigned long long stride,
                                            unsigned tilew, unsigned tileh) {
  u32x4 g0; i32x8 g1;
  tdm_build(g0, g1, ldsoff, (unsigned long long)(uintptr_t)gptr, tw, th, stride,
            tilew, tileh);
  i32x4 z4 = {0, 0, 0, 0};
#if __clang_major__ >= 23
  i32x8 z8 = {};
  __builtin_amdgcn_tensor_load_to_lds(g0, g1, z4, z4, z8, 0);
#else
  __builtin_amdgcn_tensor_load_to_lds(g0, g1, z4, z4, 0);
#endif
}

__device__ __forceinline__ void tdm_store_2d(unsigned ldsoff, void* gptr,
                                             unsigned long long tw, unsigned th,
                                             unsigned long long stride,
                                             unsigned tilew, unsigned tileh) {
  u32x4 g0; i32x8 g1;
  tdm_build(g0, g1, ldsoff, (unsigned long long)(uintptr_t)gptr, tw, th, stride,
            tilew, tileh);
  i32x4 z4 = {0, 0, 0, 0};
#if __clang_major__ >= 23
  i32x8 z8 = {};
  __builtin_amdgcn_tensor_store_from_lds(g0, g1, z4, z4, z8, 0);
#else
  __builtin_amdgcn_tensor_store_from_lds(g0, g1, z4, z4, 0);
#endif
}
#endif  // USE_TDM

// ---------------------------------------------------------------------------
// B fragment (16x16x32 f16 WMMA): lane L -> column n = ntile*16 + L%16,
// klo = (L<16)?0:8 ; VGPR j: K = klo+2j ; VGPR 4+j: K = klo+16+2j
// ---------------------------------------------------------------------------
__device__ __forceinline__ v16h load_frag_b(const _Float16* __restrict__ w,
                                            int lane, int ntile, int k0, int kstride) {
  FragH f;
  const int nn  = ntile * 16 + (lane & 15);
  const int klo = (lane < 16) ? 0 : 8;
  const _Float16* row = w + nn * kstride + k0 + klo;
#pragma unroll
  for (int j = 0; j < 4; ++j) {
    f.u[j]     = *reinterpret_cast<const unsigned*>(row + 2 * j);
    f.u[4 + j] = *reinterpret_cast<const unsigned*>(row + 16 + 2 * j);
  }
  return f.h;
}

// ---------------------------------------------------------------------------
// A fragment via on-the-fly im2col from LDS activation [CIN][IH][ISTR] f16.
// Row m = output position (oy*OW+ox), K = c*16 + ky*4 + kx (4x4 window, s2).
// ---------------------------------------------------------------------------
template<int OW, int ISTR, int CHSTR, int KVALID, bool NEEDK>
__device__ __forceinline__ v16h load_frag_a(const _Float16* __restrict__ x,
                                            int lane, int mbase, int kabs0, int validM) {
  FragH f;
  const int  m   = mbase + (lane & 15);
  const bool okm = (m < validM);
  const int  mm  = okm ? m : 0;
  const int  oy  = mm / OW, ox = mm - oy * OW;
  const int  klo = (lane < 16) ? 0 : 8;
  const _Float16* base = x + 2 * oy * ISTR + 2 * ox;
#pragma unroll
  for (int j = 0; j < 8; ++j) {
    const int  k   = kabs0 + klo + ((j < 4) ? 2 * j : 16 + 2 * (j - 4));
    const bool okk = NEEDK ? (k < KVALID) : true;
    const int  kk  = okk ? k : 0;
    const int  c = kk >> 4, r = kk & 15, ky = r >> 2, kx = r & 3;
    const unsigned v =
        *reinterpret_cast<const unsigned*>(base + c * CHSTR + ky * ISTR + kx);
    f.u[j] = (okm && okk) ? v : 0u;
  }
  return f.h;
}

// ---------------------------------------------------------------------------
// Fused conv(4x4,s2,VALID) [+instance-norm+leaky-relu] or [+mean-pool+bias].
// One workgroup (8 wave32 waves) per crop; TDM double-buffered weight chunks.
// ---------------------------------------------------------------------------
template<int CIN, int COUT, int IH, int IW, int ISTR,
         int OH, int OW, int OSTR, int KC, bool NORM>
__global__ __launch_bounds__(256) void conv_wmma_kernel(
    const _Float16* __restrict__ xin,   // [N][CIN][IH][ISTR] f16
    const _Float16* __restrict__ wpk,   // [COUT][KPAD] f16 (zero-padded K)
    const float*    __restrict__ bias,  // [COUT] (used when !NORM)
    _Float16*       __restrict__ yout,  // [N][COUT][OH][OSTR] f16 (if NORM)
    float*          __restrict__ pooled)// [N][COUT] f32 (if !NORM)
{
  constexpr int KVALID = CIN * 16;
  constexpr int KPAD   = ((KVALID + KC - 1) / KC) * KC;
  constexpr int NCHUNK = KPAD / KC;
  constexpr int M      = OH * OW;
  constexpr int MT     = (M + 15) / 16;
  constexpr int NT     = COUT / 16;
  constexpr int INE    = CIN * IH * ISTR;
  constexpr int CHSTR  = IH * ISTR;
  constexpr int KF     = KC / 32;
  constexpr int OUTE   = COUT * OH * OSTR;
  // wave -> fixed column tiles; rows strided across waves sharing a column.
  constexpr int NTW     = (NT + 7) / 8;              // n-tiles per wave
  constexpr int MSTRIDE = (NT >= 8) ? 1 : (8 / NT);  // mt stride
  constexpr int MPW     = (MT + MSTRIDE - 1) / MSTRIDE;
  static_assert(INE % 8 == 0 && (COUT * KC) % 8 == 0, "vector staging");

  extern __shared__ char smem[];
  _Float16* s_in  = reinterpret_cast<_Float16*>(smem);
  _Float16* s_w0  = s_in + INE;
  _Float16* s_w1  = s_w0 + COUT * KC;
  _Float16* s_out = s_w1 + COUT * KC;                // raw conv out (NORM)
  float*    s_sum = reinterpret_cast<float*>(s_out + (NORM ? OUTE : 0));
  float*    s_sqs = s_sum + COUT;
  float*    s_mu  = s_sqs + COUT;
  float*    s_ri  = s_mu + COUT;

  const int  n      = blockIdx.x;
  const int  tid    = threadIdx.x;
  const int  lane   = tid & 31;
  const int  wave   = tid >> 5;
  const bool issuer = (wave == 0);   // wave-uniform

#if USE_TDM
  if (issuer) {
    tdm_load_2d(lds_off(s_in), xin + (size_t)n * INE, INE, 1, INE, INE, 1);
    tdm_load_2d(lds_off(s_w0), wpk, KPAD, COUT, KPAD, KC, COUT);
  }
#else
  {
    const uint4* g = reinterpret_cast<const uint4*>(xin + (size_t)n * INE);
    uint4* s = reinterpret_cast<uint4*>(s_in);
    for (int i = tid; i < INE / 8; i += 256) s[i] = g[i];
  }
#endif
  for (int c = tid; c < COUT; c += 256) { s_sum[c] = 0.f; s_sqs[c] = 0.f; }

  float acc[NTW][MPW][8];
#pragma unroll
  for (int jn = 0; jn < NTW; ++jn)
#pragma unroll
    for (int im = 0; im < MPW; ++im)
#pragma unroll
      for (int v = 0; v < 8; ++v) acc[jn][im][v] = 0.f;

  for (int ch = 0; ch < NCHUNK; ++ch) {
#if USE_TDM
    if (issuer) __builtin_amdgcn_s_wait_tensorcnt(0);
    __syncthreads();
    if (issuer && (ch + 1 < NCHUNK))
      tdm_load_2d(lds_off(((ch + 1) & 1) ? s_w1 : s_w0),
                  wpk + (size_t)(ch + 1) * KC, KPAD, COUT, KPAD, KC, COUT);
#else
    {
      _Float16* dst = (ch & 1) ? s_w1 : s_w0;
      for (int i = tid; i < COUT * KC / 8; i += 256) {
        const int e = i * 8, nn = e / KC, kk = e - nn * KC;
        reinterpret_cast<uint4*>(dst)[i] =
            *reinterpret_cast<const uint4*>(wpk + (size_t)nn * KPAD + ch * KC + kk);
      }
      __syncthreads();
    }
#endif
    const _Float16* s_wc = (ch & 1) ? s_w1 : s_w0;

    // B fragments: loaded once per chunk per wave (fixed column tiles).
    v16h bf[NTW][KF];
#pragma unroll
    for (int jn = 0; jn < NTW; ++jn) {
      const int nt = (NT >= 8) ? (wave + 8 * jn) : (wave % NT);
#pragma unroll
      for (int kf = 0; kf < KF; ++kf)
        bf[jn][kf] = load_frag_b(s_wc, lane, nt, kf * 32, KC);
    }
#pragma unroll
    for (int im = 0; im < MPW; ++im) {
      const int mt = (NT >= 8) ? im : (wave / NT + MSTRIDE * im);
      if (mt < MT) {                    // wave-uniform: EXEC stays full
#pragma unroll
        for (int kf = 0; kf < KF; ++kf) {
          v16h a = load_frag_a<OW, ISTR, CHSTR, KVALID, (KVALID != KPAD)>(
              s_in, lane, mt * 16, ch * KC + kf * 32, M);
#pragma unroll
          for (int jn = 0; jn < NTW; ++jn) {
            v8f c8;
#pragma unroll
            for (int v = 0; v < 8; ++v) c8[v] = acc[jn][im][v];
            c8 = __builtin_amdgcn_wmma_f32_16x16x32_f16(
                false, a, false, bf[jn][kf], (short)0, c8, false, false);
#pragma unroll
            for (int v = 0; v < 8; ++v) acc[jn][im][v] = c8[v];
          }
        }
      }
    }
#if !USE_TDM
    __syncthreads();
#endif
  }

  // Epilogue: D lane layout -> column = lane%16, rows = mt*16+v+(lane<16?0:8)
#pragma unroll
  for (int im = 0; im < MPW; ++im) {
    const int mt = (NT >= 8) ? im : (wave / NT + MSTRIDE * im);
    if (mt < MT) {
#pragma unroll
      for (int jn = 0; jn < NTW; ++jn) {
        const int nt   = (NT >= 8) ? (wave + 8 * jn) : (wave % NT);
        const int ncol = nt * 16 + (lane & 15);
        const int hi8  = (lane < 16) ? 0 : 8;
        float ps = 0.f, pq = 0.f;
#pragma unroll
        for (int v = 0; v < 8; ++v) {
          const int   m   = mt * 16 + v + hi8;
          const float val = acc[jn][im][v];
          if (m < M) {
            if (NORM) {
              const int oy = m / OW, ox = m - oy * OW;
              s_out[ncol * OH * OSTR + oy * OSTR + ox] = (_Float16)val;
            }
            ps += val;
            pq += val * val;
          }
        }
        atomicAdd(&s_sum[ncol], ps);
        if (NORM) atomicAdd(&s_sqs[ncol], pq);
      }
    }
  }
  __syncthreads();

  if (NORM) {
    for (int c = tid; c < COUT; c += 256) {
      const float mu  = s_sum[c] / (float)M;
      const float var = s_sqs[c] / (float)M - mu * mu;
      s_mu[c] = mu;
      s_ri[c] = rsqrtf(var + 1e-5f);
    }
    __syncthreads();
    for (int idx = tid; idx < COUT * M; idx += 256) {
      const int c = idx / M, rm = idx - c * M;
      const int oy = rm / OW, ox = rm - oy * OW;
      const int si = c * OH * OSTR + oy * OSTR + ox;
      float x = (float)s_out[si];
      x = (x - s_mu[c]) * s_ri[c];
      x = (x >= 0.f) ? x : 0.2f * x;
      s_out[si] = (_Float16)x;
    }
    __syncthreads();
#if USE_TDM
    if (issuer)   // S_ENDPGM implies wait-idle (covers TENSORcnt)
      tdm_store_2d(lds_off(s_out), yout + (size_t)n * OUTE, OUTE, 1, OUTE, OUTE, 1);
#else
    {
      const uint4* s = reinterpret_cast<const uint4*>(s_out);
      uint4* g = reinterpret_cast<uint4*>(yout + (size_t)n * OUTE);
      for (int i = tid; i < OUTE / 8; i += 256) g[i] = s[i];
    }
#endif
  } else {
    for (int c = tid; c < COUT; c += 256)
      pooled[(size_t)n * COUT + c] = s_sum[c] / (float)M + bias[c];
  }
}

// ---------------------------------------------------------------------------
// Bilinear crop of 512 boxes -> (512,3,64,64) f16 (border-masked taps).
// ---------------------------------------------------------------------------
__global__ __launch_bounds__(256) void crop_kernel(
    const float* __restrict__ img, const float* __restrict__ boxes,
    _Float16* __restrict__ crops) {
  const int n = blockIdx.x, tid = threadIdx.x;
  const float bx0 = 2.f * boxes[n * 4 + 0] - 1.f;
  const float by0 = 2.f * boxes[n * 4 + 1] - 1.f;
  const float bx1 = 2.f * boxes[n * 4 + 2] - 1.f;
  const float by1 = 2.f * boxes[n * 4 + 3] - 1.f;
  for (int idx = tid; idx < 64 * 64; idx += 256) {
    const int i = idx >> 6, j = idx & 63;
    const float tj = (float)j * (1.f / 63.f), ti = (float)i * (1.f / 63.f);
    const float X = bx0 * (1.f - tj) + bx1 * tj;
    const float Y = by0 * (1.f - ti) + by1 * ti;
    const float ix = ((X + 1.f) * 512.f - 1.f) * 0.5f;
    const float iy = ((Y + 1.f) * 512.f - 1.f) * 0.5f;
    const float x0f = floorf(ix), y0f = floorf(iy);
    const float wx = ix - x0f, wy = iy - y0f;
    const int x0 = (int)x0f, y0 = (int)y0f;
    const int x1 = x0 + 1, y1 = y0 + 1;
    const float mx0 = (x0 >= 0 && x0 <= 511) ? 1.f : 0.f;
    const float mx1 = (x1 >= 0 && x1 <= 511) ? 1.f : 0.f;
    const float my0 = (y0 >= 0 && y0 <= 511) ? 1.f : 0.f;
    const float my1 = (y1 >= 0 && y1 <= 511) ? 1.f : 0.f;
    const int cx0 = x0 < 0 ? 0 : (x0 > 511 ? 511 : x0);
    const int cx1 = x1 < 0 ? 0 : (x1 > 511 ? 511 : x1);
    const int cy0 = y0 < 0 ? 0 : (y0 > 511 ? 511 : y0);
    const int cy1 = y1 < 0 ? 0 : (y1 > 511 ? 511 : y1);
    const float w00 = (1.f - wx) * (1.f - wy) * mx0 * my0;
    const float w10 = wx * (1.f - wy) * mx1 * my0;
    const float w01 = (1.f - wx) * wy * mx0 * my1;
    const float w11 = wx * wy * mx1 * my1;
#pragma unroll
    for (int c = 0; c < 3; ++c) {
      const float* ic = img + c * 262144;
      const float v = ic[cy0 * 512 + cx0] * w00 + ic[cy0 * 512 + cx1] * w10 +
                      ic[cy1 * 512 + cx0] * w01 + ic[cy1 * 512 + cx1] * w11;
      crops[(size_t)n * 12288 + c * 4096 + idx] = (_Float16)v;
    }
  }
}

// ---------------------------------------------------------------------------
__global__ void pack_weights(const float* __restrict__ src,
                             _Float16* __restrict__ dst,
                             int N, int CIN, int KPAD) {
  const int idx = blockIdx.x * 256 + threadIdx.x;
  if (idx >= N * KPAD) return;
  const int nn = idx / KPAD, k = idx - nn * KPAD;
  float v = 0.f;
  if (k < CIN * 16) {
    const int c = k >> 4, r = k & 15, ky = r >> 2, kx = r & 3;
    v = src[((nn * CIN + c) * 4 + ky) * 4 + kx];
  }
  dst[idx] = (_Float16)v;
}

__global__ void zero_loss(float* out) {
  if (blockIdx.x == 0 && threadIdx.x == 0) out[512] = 0.f;
}

// ---------------------------------------------------------------------------
__global__ __launch_bounds__(256) void head_kernel(
    const float* __restrict__ pooled, const int* __restrict__ objs,
    const float* __restrict__ lin_w, const float* __restrict__ lin_b,
    const float* __restrict__ real_w, const float* __restrict__ real_b,
    const float* __restrict__ cls_w, const float* __restrict__ cls_b,
    float* __restrict__ out) {
  __shared__ float s_p[256];
  __shared__ float s_v[512];
  __shared__ float s_r[256];
  __shared__ float s_l[180];
  const int n = blockIdx.x, tid = threadIdx.x;
  s_p[tid] = pooled[(size_t)n * 256 + tid];
  __syncthreads();
#pragma unroll
  for (int oo = 0; oo < 2; ++oo) {
    const int o = tid + oo * 256;
    float s = lin_b[o];
    for (int k = 0; k < 256; ++k) s += s_p[k] * lin_w[o * 256 + k];
    s_v[o] = s;
  }
  __syncthreads();
  s_r[tid] = s_v[tid] * real_w[tid] + s_v[tid + 256] * real_w[tid + 256];
  __syncthreads();
  for (int st = 128; st > 0; st >>= 1) {
    if (tid < st) s_r[tid] += s_r[tid + st];
    __syncthreads();
  }
  if (tid == 0) {
    const float z = s_r[0] + real_b[0];
    out[n] = 1.f / (1.f + expf(-z));
  }
  if (tid < 180) {
    float s = cls_b[tid];
    for (int k = 0; k < 512; ++k) s += s_v[k] * cls_w[tid * 512 + k];
    s_l[tid] = s;
  }
  __syncthreads();
  if (tid == 0) {
    float mx = s_l[0];
    for (int i = 1; i < 180; ++i) mx = fmaxf(mx, s_l[i]);
    float se = 0.f;
    for (int i = 0; i < 180; ++i) se += expf(s_l[i] - mx);
    const float lse = mx + logf(se);
    const float lp = s_l[objs[n]] - lse;
    atomicAdd(&out[512], -lp * (0.1f / 512.f));
  }
}

// ---------------------------------------------------------------------------
extern "C" void kernel_launch(void* const* d_in, const int* in_sizes, int n_in,
                              void* d_out, int out_size, void* d_ws, size_t ws_size,
                              hipStream_t stream) {
  const float* img    = (const float*)d_in[0];
  const int*   objs   = (const int*)d_in[1];
  const float* boxes  = (const float*)d_in[2];
  const float* w1     = (const float*)d_in[3];
  const float* b1     = (const float*)d_in[4];
  const float* w2     = (const float*)d_in[5];
  const float* b2     = (const float*)d_in[6];
  const float* w3     = (const float*)d_in[7];
  const float* b3     = (const float*)d_in[8];
  const float* lin_w  = (const float*)d_in[9];
  const float* lin_b  = (const float*)d_in[10];
  const float* real_w = (const float*)d_in[11];
  const float* real_b = (const float*)d_in[12];
  const float* cls_w  = (const float*)d_in[13];
  const float* cls_b  = (const float*)d_in[14];
  float* out = (float*)d_out;

  char* ws = (char*)d_ws;
  size_t off = 0;
  auto carve = [&](size_t bytes) -> char* {
    char* p = ws + off;
    off = (off + bytes + 255) & ~(size_t)255;
    return p;
  };
  _Float16* crops  = (_Float16*)carve(512ull * 3 * 64 * 64 * 2);
  _Float16* c1     = (_Float16*)carve(512ull * 64 * 31 * 32 * 2);
  _Float16* c2     = (_Float16*)carve(512ull * 128 * 14 * 16 * 2);
  float*    pooled = (float*)   carve(512ull * 256 * 4);
  _Float16* w1p    = (_Float16*)carve(64ull * 64 * 2);
  _Float16* w2p    = (_Float16*)carve(128ull * 1024 * 2);
  _Float16* w3p    = (_Float16*)carve(256ull * 2048 * 2);

  zero_loss<<<1, 32, 0, stream>>>(out);
  pack_weights<<<(64 * 64 + 255) / 256, 256, 0, stream>>>(w1, w1p, 64, 3, 64);
  pack_weights<<<(128 * 1024 + 255) / 256, 256, 0, stream>>>(w2, w2p, 128, 64, 1024);
  pack_weights<<<(256 * 2048 + 255) / 256, 256, 0, stream>>>(w3, w3p, 256, 128, 2048);
  crop_kernel<<<512, 256, 0, stream>>>(img, boxes, crops);

  // conv1: 3->64, in 64x64 (stride 64), out 31x31 (stride 32), K=48 pad 64
  conv_wmma_kernel<3, 64, 64, 64, 64, 31, 31, 32, 64, true>
      <<<512, 256, 168960, stream>>>(crops, w1p, b1, c1, nullptr);
  // conv2: 64->128, in 31x31 (stride 32), out 14x14 (stride 16), K=1024
  conv_wmma_kernel<64, 128, 31, 31, 32, 14, 14, 16, 128, true>
      <<<512, 256, 251904, stream>>>(c1, w2p, b2, c2, nullptr);
  // conv3: 128->256, in 14x14 (stride 16), out 6x6, K=2048, mean-pool + b3
  conv_wmma_kernel<128, 256, 14, 14, 16, 6, 6, 8, 128, false>
      <<<512, 256, 192512, stream>>>(c2, w3p, b3, nullptr, pooled);

  head_kernel<<<512, 256, 0, stream>>>(pooled, objs, lin_w, lin_b,
                                       real_w, real_b, cls_w, cls_b, out);
}
